// Twopath_NonLocalBlockND_32478542692409
// MI455X (gfx1250) — matrix operand order repaired and checked by reading, not compiled, not run on hardware
//
#include <hip/hip_runtime.h>

// ---------------- problem constants ----------------
#define B_   16
#define CC   20
#define CG   256
#define CI   10
#define CIG  128
#define H_   80
#define W_   80
#define N_   6400   // 80*80
#define NP   1600   // 40*40

typedef __attribute__((ext_vector_type(16))) _Float16 v16h;
typedef __attribute__((ext_vector_type(8)))  float    v8f;

// ---------------- WMMA helpers (wave32, 16x16x32 f16 -> f32) ----------------
__device__ __forceinline__ v8f wmma16(v16h a, v16h b, v8f c) {
  return __builtin_amdgcn_wmma_f32_16x16x32_f16(false, a, false, b, (short)0, c, false, false);
}

// A fragment 16x32 from row-major f16, row stride ld (halves), tile origin (row0,k0).
// Lanes 0-15 -> K {0..7,16..23}; lanes 16-31 -> K {8..15,24..31}.
__device__ __forceinline__ v16h load_a(const _Float16* base, int ld, int row0, int k0) {
  int l = threadIdx.x & 31;
  const _Float16* p = base + (size_t)(row0 + (l & 15)) * ld + k0 + ((l >> 4) * 8);
  v16h a;
#pragma unroll
  for (int i = 0; i < 8; ++i) { a[i] = p[i]; a[8 + i] = p[16 + i]; }
  return a;
}

// A fragment where only K=0..15 is valid (upper half of K zero-padded), branch-free
__device__ __forceinline__ v16h load_a_k16(const _Float16* base, int ld, int row0) {
  int l = threadIdx.x & 31;
  const _Float16* p = base + (size_t)(row0 + (l & 15)) * ld + ((l >> 4) * 8);
  v16h a;
#pragma unroll
  for (int i = 0; i < 8; ++i) { a[i] = p[i]; a[8 + i] = (_Float16)0.f; }
  return a;
}

// A fragment from an f32 LDS buffer (rows 0..15 of `base`), converting to f16
__device__ __forceinline__ v16h load_a_lds(const float* base, int ld, int k0) {
  int l = threadIdx.x & 31;
  const float* p = base + (size_t)(l & 15) * ld + k0 + ((l >> 4) * 8);
  v16h a;
#pragma unroll
  for (int i = 0; i < 8; ++i) { a[i] = (_Float16)p[i]; a[8 + i] = (_Float16)p[16 + i]; }
  return a;
}

// B fragment 32x16: each column's K contiguous (column n at base + n*ld).
// Lanes 0-15 hold K {0..7,16..23} of col n, lanes 16-31 hold K {8..15,24..31}.
__device__ __forceinline__ v16h load_b(const _Float16* base, int ld, int n0, int k0) {
  int l = threadIdx.x & 31;
  const _Float16* p = base + (size_t)(n0 + (l & 15)) * ld + k0 + ((l >> 4) * 8);
  v16h b;
#pragma unroll
  for (int i = 0; i < 8; ++i) { b[i] = p[i]; b[8 + i] = p[16 + i]; }
  return b;
}

// ---------------- prep kernels ----------------
// theta: (B, N, 16) f16, K padded 10->16 with zeros, bias folded in
__global__ void k_theta(const float* __restrict__ xc, const float* __restrict__ tw,
                        const float* __restrict__ tb, _Float16* __restrict__ theta) {
  int idx = blockIdx.x * blockDim.x + threadIdx.x;
  if (idx >= B_ * N_ * 16) return;
  int k = idx & 15; int n = (idx >> 4) % N_; int b = idx / (16 * N_);
  float acc = 0.f;
  if (k < CI) {
    acc = tb[k];
    const float* x = xc + (size_t)b * CC * N_ + n;
#pragma unroll
    for (int c = 0; c < CC; ++c) acc += x[(size_t)c * N_] * tw[k * CC + c];
  }
  theta[idx] = (_Float16)acc;
}

// phi: (B, Np, 32) f16, column-contiguous K padded 10->32 (branch-free B loads)
__global__ void k_phi(const float* __restrict__ xc, const float* __restrict__ pw,
                      const float* __restrict__ pb, _Float16* __restrict__ phi) {
  int idx = blockIdx.x * blockDim.x + threadIdx.x;
  if (idx >= B_ * NP * 32) return;
  int k = idx & 31; int m = (idx >> 5) % NP; int b = idx / (32 * NP);
  float r = 0.f;
  if (k < CI) {
    int my = m / 40, mx = m % 40;
    const float* x = xc + (size_t)b * CC * N_;
    float best = -3.4e38f;
#pragma unroll
    for (int dy = 0; dy < 2; ++dy)
#pragma unroll
      for (int dx = 0; dx < 2; ++dx) {
        int n = (2 * my + dy) * W_ + (2 * mx + dx);
        float acc = pb[k];
#pragma unroll
        for (int c = 0; c < CC; ++c) acc += x[(size_t)c * N_ + n] * pw[k * CC + c];
        best = fmaxf(best, acc);
      }
    r = best;
  }
  phi[idx] = (_Float16)r;
}

// f32 -> f16 conversion (used for W_w and g_w)
__global__ void k_cvt(const float* __restrict__ src, _Float16* __restrict__ dst, int n) {
  int idx = blockIdx.x * blockDim.x + threadIdx.x;
  if (idx < n) dst[idx] = (_Float16)src[idx];
}

__global__ void k_zero(float* g) {
  int t = blockIdx.x * blockDim.x + threadIdx.x;
  if (t < 512) g[t] = 0.f;
}

// ---------------- g-path conv as WMMA GEMM ----------------
// convg(B, N, 128) f16 = x_1(B, 256, N)^T @ g_w(128,256)^T  (bias deferred to pool)
#define GLD 264   // padded LDS row stride (halves) to break bank-conflict alignment
__global__ __launch_bounds__(256) void k_gconv(const float* __restrict__ x1,
                                               const _Float16* __restrict__ gwh,
                                               _Float16* __restrict__ convg) {
  __shared__ _Float16 sx[64 * GLD];   // 64 rows (n) x 256 ch, padded: ~33.8 KB
  int b = blockIdx.y;
  int n0 = blockIdx.x * 64;
  int t = threadIdx.x, w = t >> 5, l = t & 31;

  const float* x = x1 + (size_t)b * CG * N_ + n0;
  for (int j = t; j < 64 * CG; j += 256) {
    int c = j >> 6, n = j & 63;              // consecutive t -> consecutive n: coalesced
    sx[n * GLD + c] = (_Float16)x[(size_t)c * N_ + n];
  }
  __syncthreads();

  int rt = w & 3;            // row tile 0..3 (16 rows each)
  int cg = w >> 2;           // col half: 64 cols each
  v16h af[8];
#pragma unroll
  for (int ks = 0; ks < 8; ++ks) af[ks] = load_a(sx, GLD, rt * 16, ks * 32);

  int nl = l & 15, rb = (l >> 4) * 8;
  _Float16* outp = convg + ((size_t)b * N_ + n0 + rt * 16) * CIG;
#pragma unroll
  for (int ct = 0; ct < 4; ++ct) {
    int c0 = cg * 64 + ct * 16;
    v8f acc = {};
#pragma unroll
    for (int ks = 0; ks < 8; ++ks)
      acc = wmma16(af[ks], load_b(gwh, CG, c0, ks * 32), acc);
#pragma unroll
    for (int i = 0; i < 8; ++i)
      outp[(size_t)(rb + i) * CIG + c0 + nl] = (_Float16)acc[i];
  }
}

// 2x2 maxpool of convg + bias -> gT (B, 128, Np) f16 (K-contiguous for y-GEMM B operand)
__global__ void k_gpool(const _Float16* __restrict__ convg, const float* __restrict__ gb,
                        _Float16* __restrict__ gT) {
  int idx = blockIdx.x * blockDim.x + threadIdx.x;
  if (idx >= B_ * CIG * NP) return;
  int c = idx & 127; int m = (idx >> 7) % NP; int b = idx / (128 * NP);
  int my = m / 40, mx = m % 40;
  int n = (2 * my) * W_ + 2 * mx;
  const _Float16* p = convg + ((size_t)b * N_ + n) * CIG + c;
  float v0 = (float)p[0];
  float v1 = (float)p[CIG];
  float v2 = (float)p[(size_t)W_ * CIG];
  float v3 = (float)p[(size_t)(W_ + 1) * CIG];
  float r = fmaxf(fmaxf(v0, v1), fmaxf(v2, v3)) + gb[c];
  gT[((size_t)b * CIG + c) * NP + m] = (_Float16)r;
}

// ---------------- fused attention kernel ----------------
// Block = (batch, 32-row tile of N), 256 threads = 8 waves.
__global__ __launch_bounds__(256) void k_attn(const _Float16* __restrict__ theta,
                                              const _Float16* __restrict__ phi,
                                              const _Float16* __restrict__ gT,
                                              _Float16* __restrict__ Y) {
  __shared__ float sf[32 * NP];     // 204800 B score/exp tile
  __shared__ float s_red[32 * 8];
  __shared__ float s_rowmax[32];
  __shared__ float s_rinv[32];

  int b = blockIdx.y;
  int row0 = blockIdx.x * 32;
  int t = threadIdx.x, w = t >> 5, l = t & 31;

  const _Float16* th = theta + (size_t)b * N_ * 16;
  const _Float16* ph = phi + (size_t)b * NP * 32;
  const _Float16* g  = gT  + (size_t)b * CIG * NP;

  // Phase 1: scores. Wave w: row-tile (w&1), m-tiles (w>>1) step 4.
  {
    int rt = w & 1;
    v16h a = load_a_k16(th, 16, row0 + rt * 16);
    for (int mt = (w >> 1); mt < NP / 16; mt += 4) {
      v16h bf = load_b(ph, 32, mt * 16, 0);   // padded K: branch-free
      v8f acc = {};
      acc = wmma16(a, bf, acc);
      int n = (l & 15) + mt * 16;
      int rb = rt * 16 + (l >> 4) * 8;
#pragma unroll
      for (int i = 0; i < 8; ++i) sf[(size_t)(rb + i) * NP + n] = acc[i];
    }
  }
  __syncthreads();

  // Phase 2: softmax (256 threads: row = t&31, 8 chunks of 200 along m)
  {
    int row = t & 31, chunk = t >> 5;
    float* fr = sf + (size_t)row * NP + chunk * 200;
    float mx = -3.4e38f;
    for (int i = 0; i < 200; ++i) mx = fmaxf(mx, fr[i]);
    s_red[row * 8 + chunk] = mx;
    __syncthreads();
    if (t < 32) {
      float mm = s_red[t * 8];
#pragma unroll
      for (int i = 1; i < 8; ++i) mm = fmaxf(mm, s_red[t * 8 + i]);
      s_rowmax[t] = mm;
    }
    __syncthreads();
    float rm = s_rowmax[row];
    float s = 0.f;
    for (int i = 0; i < 200; ++i) { float e = __expf(fr[i] - rm); fr[i] = e; s += e; }
    s_red[row * 8 + chunk] = s;
    __syncthreads();
    if (t < 32) {
      float ss = 0.f;
#pragma unroll
      for (int i = 0; i < 8; ++i) ss += s_red[t * 8 + i];
      s_rinv[t] = 1.f / ss;
    }
    __syncthreads();
  }

  // Phase 3: y(32x128) = P(32x1600) @ g(1600x128).
  {
    int rt = w & 1;
    int c0 = (w >> 1) * 32;
    const float* sfr = sf + (size_t)(rt * 16) * NP;
    v8f acc0 = {}, acc1 = {};
    const _Float16* gc0 = g + (size_t)(c0 + (l & 15)) * NP;
    for (int ks = 0; ks < NP / 32; ++ks) {
      int k0 = ks * 32;
      __builtin_prefetch(gc0 + k0 + 64, 0, 1);   // global_prefetch_b8 on the g stream
      v16h pa = load_a_lds(sfr, NP, k0);
      v16h b0 = load_b(g, NP, c0, k0);
      acc0 = wmma16(pa, b0, acc0);
      v16h b1 = load_b(g, NP, c0 + 16, k0);
      acc1 = wmma16(pa, b1, acc1);
    }
    int n = l & 15, rb = rt * 16 + (l >> 4) * 8;
    _Float16* yout = Y + ((size_t)b * N_ + row0) * CIG;
#pragma unroll
    for (int i = 0; i < 8; ++i) {
      float rv = s_rinv[rb + i];
      yout[(size_t)(rb + i) * CIG + c0 + n]      = (_Float16)(acc0[i] * rv);
      yout[(size_t)(rb + i) * CIG + c0 + 16 + n] = (_Float16)(acc1[i] * rv);
    }
  }
}

// ---------------- w_y = Y @ W^T + b, with BN-stat accumulation ----------------
__global__ __launch_bounds__(128) void k_wconv(const _Float16* __restrict__ Y,
                                               const _Float16* __restrict__ Wh,
                                               const float* __restrict__ Wb,
                                               float* __restrict__ WY,
                                               float* __restrict__ gstats) {
  __shared__ float s_sum[CG], s_sq[CG];
  int t = threadIdx.x, w = t >> 5, l = t & 31;
  for (int i = t; i < CG; i += 128) { s_sum[i] = 0.f; s_sq[i] = 0.f; }
  __syncthreads();

  size_t row0 = (size_t)blockIdx.x * 64 + (size_t)w * 16;
  const _Float16* ab = Y + row0 * CIG;
  v16h a0 = load_a(ab, CIG, 0, 0);
  v16h a1 = load_a(ab, CIG, 0, 32);
  v16h a2 = load_a(ab, CIG, 0, 64);
  v16h a3 = load_a(ab, CIG, 0, 96);

  int n = l & 15, rb = (l >> 4) * 8;
  for (int ct = 0; ct < CG / 16; ++ct) {
    int c0 = ct * 16;
    v8f acc = {};
    acc = wmma16(a0, load_b(Wh, CIG, c0, 0), acc);
    acc = wmma16(a1, load_b(Wh, CIG, c0, 32), acc);
    acc = wmma16(a2, load_b(Wh, CIG, c0, 64), acc);
    acc = wmma16(a3, load_b(Wh, CIG, c0, 96), acc);
    int o = c0 + n;
    float bias = Wb[o];
    float ps = 0.f, pq = 0.f;
    float* outp = WY + row0 * CG + o;
#pragma unroll
    for (int i = 0; i < 8; ++i) {
      float v = acc[i] + bias;
      outp[(size_t)(rb + i) * CG] = v;
      ps += v; pq += v * v;
    }
    atomicAdd(&s_sum[o], ps);
    atomicAdd(&s_sq[o], pq);
  }
  __syncthreads();
  for (int i = t; i < CG; i += 128) {
    atomicAdd(&gstats[i], s_sum[i]);
    atomicAdd(&gstats[CG + i], s_sq[i]);
  }
}

// ---------------- BN parameter fold: scale/shift per channel ----------------
__global__ void k_bnparm(const float* __restrict__ gstats, const float* __restrict__ gamma,
                         const float* __restrict__ beta, float* __restrict__ bnp) {
  int o = blockIdx.x * blockDim.x + threadIdx.x;
  if (o >= CG) return;
  const float cnt = (float)(B_ * N_);
  float mean = gstats[o] / cnt;
  float var  = gstats[CG + o] / cnt - mean * mean;
  float sc = gamma[o] * rsqrtf(var + 1e-5f);
  bnp[o] = sc;
  bnp[CG + o] = beta[o] - mean * sc;
}

// ---------------- finalize: tiled transpose + BN + residual ----------------
// WY is (B, N, 256); out/x_1 are (B, 256, N). 64x64 LDS transpose tile keeps
// both the WY reads and the out/x_1 accesses fully coalesced.
__global__ __launch_bounds__(256) void k_final(const float* __restrict__ WY,
                                               const float* __restrict__ bnp,
                                               const float* __restrict__ x1,
                                               float* __restrict__ out) {
  __shared__ float tile[64 * 65];
  int b = blockIdx.z;
  int n0 = blockIdx.x * 64;
  int o0 = blockIdx.y * 64;
  int t = threadIdx.x;

  const float* src = WY + ((size_t)b * N_ + n0) * CG + o0;
  int j = t & 63;                       // o within tile: contiguous -> coalesced
  for (int i = t >> 6; i < 64; i += 4)  // n within tile
    tile[i * 65 + j] = src[(size_t)i * CG + j];
  __syncthreads();

  int n = t & 63;                       // n contiguous -> coalesced out/x_1
  for (int o = t >> 6; o < 64; o += 4) {
    float sc = bnp[o0 + o];
    float sh = bnp[CG + o0 + o];
    size_t oidx = (size_t)b * CG * N_ + (size_t)(o0 + o) * N_ + n0 + n;
    out[oidx] = tile[n * 65 + o] * sc + sh + x1[oidx];
  }
}

// ---------------- launcher ----------------
extern "C" void kernel_launch(void* const* d_in, const int* in_sizes, int n_in,
                              void* d_out, int out_size, void* d_ws, size_t ws_size,
                              hipStream_t stream) {
  const float* xc    = (const float*)d_in[0];
  const float* x1    = (const float*)d_in[1];
  const float* tw    = (const float*)d_in[2];
  const float* tb    = (const float*)d_in[3];
  const float* pw    = (const float*)d_in[4];
  const float* pb    = (const float*)d_in[5];
  const float* gw    = (const float*)d_in[6];
  const float* gb    = (const float*)d_in[7];
  const float* Ww    = (const float*)d_in[8];
  const float* Wb    = (const float*)d_in[9];
  const float* gamma = (const float*)d_in[10];
  const float* beta  = (const float*)d_in[11];

  char* ws = (char*)d_ws;
  size_t off = 0;
  auto alloc = [&](size_t bytes) -> void* {
    void* p = ws + off;
    off += (bytes + 255) & ~(size_t)255;
    return p;
  };
  _Float16* theta = (_Float16*)alloc((size_t)B_ * N_ * 16 * 2);
  _Float16* phi   = (_Float16*)alloc((size_t)B_ * NP * 32 * 2);
  _Float16* gT    = (_Float16*)alloc((size_t)B_ * CIG * NP * 2);
  _Float16* Wh    = (_Float16*)alloc((size_t)CG * CIG * 2);
  _Float16* gwh   = (_Float16*)alloc((size_t)CIG * CG * 2);
  _Float16* convg = (_Float16*)alloc((size_t)B_ * N_ * CIG * 2);
  _Float16* Yb    = (_Float16*)alloc((size_t)B_ * N_ * CIG * 2);
  float*    WY    = (float*)alloc((size_t)B_ * N_ * CG * 4);
  float*    gstats= (float*)alloc((size_t)2 * CG * 4);
  float*    bnp   = (float*)alloc((size_t)2 * CG * 4);

  // prep
  k_theta<<<(B_ * N_ * 16 + 255) / 256, 256, 0, stream>>>(xc, tw, tb, theta);
  k_phi<<<(B_ * NP * 32 + 255) / 256, 256, 0, stream>>>(xc, pw, pb, phi);
  k_cvt<<<(CG * CIG + 255) / 256, 256, 0, stream>>>(Ww, Wh, CG * CIG);
  k_cvt<<<(CIG * CG + 255) / 256, 256, 0, stream>>>(gw, gwh, CIG * CG);
  k_zero<<<2, 256, 0, stream>>>(gstats);

  // g path: WMMA conv then pool
  dim3 gg(N_ / 64, B_);
  k_gconv<<<gg, 256, 0, stream>>>(x1, gwh, convg);
  k_gpool<<<(B_ * CIG * NP + 255) / 256, 256, 0, stream>>>(convg, gb, gT);

  // fused attention
  dim3 ga(N_ / 32, B_);
  k_attn<<<ga, 256, 0, stream>>>(theta, phi, gT, Yb);

  // output conv + BN stats
  k_wconv<<<(B_ * N_) / 64, 128, 0, stream>>>(Yb, Wh, Wb, WY, gstats);

  // fold BN params, then tiled finalize
  k_bnparm<<<1, 256, 0, stream>>>(gstats, gamma, beta, bnp);
  dim3 gf(N_ / 64, CG / 64, B_);
  k_final<<<gf, 256, 0, stream>>>(WY, bnp, x1, (float*)d_out);
}
